// TransformerDecoder_80917183856673
// MI455X (gfx1250) — matrix operand order repaired and checked by reading, not compiled, not run on hardware
//
#include <hip/hip_runtime.h>
#include <math.h>

typedef __attribute__((ext_vector_type(16))) _Float16 v16h;
typedef __attribute__((ext_vector_type(8)))  float    v8f;

#define NLAY 4
#define NH   8
#define DMOD 512
#define DKH  64
#define HD   (NH*DKH)   // 512
#define DFFN 2048
#define NVOC 32000
#define NBAT 2
#define SLEN 1024
#define BLTOT (NBAT*SLEN) // 2048

// ---------------- WMMA fragment loaders (per CDNA5 ISA 7.12.2 layouts) ----------------
// A fragment (16x32 f16): lane holds row = lane&15; elements 0..7 -> k = half*8+e,
// elements 8..15 -> k = 16+half*8+(e-8). Two contiguous 16B loads.
__device__ __forceinline__ v16h frag_a(const _Float16* p, int hl) {
    union { uint4 u[2]; v16h h; } t;
    t.u[0] = *(const uint4*)(p + hl * 8);
    t.u[1] = *(const uint4*)(p + 16 + hl * 8);
    return t.h;
}
// B fragment (32x16 f16) read from a transposed [N,K] layout: lane holds col = lane&15;
// elements 0..7 -> k = half*8+e, elements 8..15 -> k = 16+half*8+(e-8). Two 16B loads.
__device__ __forceinline__ v16h frag_b(const _Float16* prow_k0, int hl) {
    union { uint4 u[2]; v16h h; } t;
    t.u[0] = *(const uint4*)(prow_k0 + hl * 8);
    t.u[1] = *(const uint4*)(prow_k0 + 16 + hl * 8);
    return t.h;
}

// ---------------- CDNA5 async global->LDS copy (ASYNCcnt-tracked) ----------------
// generic addr of a __shared__ object carries the LDS
// offset in its low 32 bits (flat-LDS aperture translation truncates to addr[31:0]).
__device__ __forceinline__ uint32_t lds_off(const void* p) {
    return (uint32_t)(uintptr_t)p;
}
__device__ __forceinline__ void async_copy_b128(uint32_t ldsoff, const void* gaddr) {
    asm volatile("global_load_async_to_lds_b128 %0, %1, off"
                 :: "v"(ldsoff), "v"(gaddr) : "memory");
}
__device__ __forceinline__ void wait_async_le3() {
    asm volatile("s_wait_asynccnt 0x3" ::: "memory");
}
__device__ __forceinline__ void wait_async_0() {
    asm volatile("s_wait_asynccnt 0x0" ::: "memory");
}

// ---------------- weight prep kernels ----------------
__global__ __launch_bounds__(256) void conv_f16_kernel(const float* __restrict__ src,
                                                       _Float16* __restrict__ dst, size_t n) {
    size_t idx = (size_t)blockIdx.x * 256 + threadIdx.x;
    if (idx < n) dst[idx] = (_Float16)src[idx];
}

// src f32 [slab][R][C]  ->  dst f16 [slab][C][R]   (slab = blockIdx.y)
__global__ __launch_bounds__(256) void convT_kernel(const float* __restrict__ src,
                                                    _Float16* __restrict__ dst, int R, int C) {
    size_t slab = blockIdx.y;
    src += slab * (size_t)R * C;
    dst += slab * (size_t)R * C;
    int idx = blockIdx.x * 256 + threadIdx.x;
    if (idx < R * C) {
        int r = idx / C, c = idx % C;
        dst[(size_t)c * R + r] = (_Float16)src[idx];
    }
}

// v16 [B, L, H, DK] f16 -> vT [B, H, DK, L] f16
__global__ __launch_bounds__(256) void transposeV_kernel(const _Float16* __restrict__ v16,
                                                         _Float16* __restrict__ vT) {
    int idx = blockIdx.x * 256 + threadIdx.x; // NBAT*NH*DKH*SLEN = 1,048,576
    int j = idx & (SLEN - 1);
    int rest = idx >> 10;
    int d = rest & (DKH - 1); rest >>= 6;
    int hh = rest & (NH - 1);
    int bb = rest >> 3;
    vT[idx] = v16[((size_t)((bb * SLEN + j) * NH + hh)) * DKH + d];
}

// ---------------- embedding + sinusoidal PE ----------------
__global__ __launch_bounds__(256) void embed_kernel(const int* __restrict__ tokens,
                                                    const float* __restrict__ emb,
                                                    float* __restrict__ x) {
    int bl = blockIdx.x;          // 0..BLTOT-1
    int i = threadIdx.x;          // pair index 0..255  (D/2)
    int tok = tokens[bl];
    int l = bl & (SLEN - 1);
    float div = __expf(-(float)(2 * i) * (logf(10000.0f) / (float)DMOD));
    float ang = (float)l * div;
    const float s = 22.62741699796952f; // sqrt(512)
    x[(size_t)bl * DMOD + 2 * i]     = emb[(size_t)tok * DMOD + 2 * i]     * s + sinf(ang);
    x[(size_t)bl * DMOD + 2 * i + 1] = emb[(size_t)tok * DMOD + 2 * i + 1] * s + cosf(ang);
}

// ---------------- LayerNorm: f32 in -> f16 out ----------------
__global__ __launch_bounds__(256) void ln_kernel(const float* __restrict__ x,
                                                 const float* __restrict__ g,
                                                 const float* __restrict__ b,
                                                 _Float16* __restrict__ out) {
    __shared__ float red[256];
    int row = blockIdx.x, tid = threadIdx.x;
    const float* xr = x + (size_t)row * DMOD;
    float a0 = xr[tid], a1 = xr[tid + 256];
    red[tid] = a0 + a1;
    __syncthreads();
    for (int s = 128; s > 0; s >>= 1) { if (tid < s) red[tid] += red[tid + s]; __syncthreads(); }
    float mean = red[0] * (1.0f / DMOD);
    __syncthreads();
    float d0 = a0 - mean, d1 = a1 - mean;
    red[tid] = d0 * d0 + d1 * d1;
    __syncthreads();
    for (int s = 128; s > 0; s >>= 1) { if (tid < s) red[tid] += red[tid + s]; __syncthreads(); }
    float inv = rsqrtf(red[0] * (1.0f / DMOD) + 1e-5f);
    out[(size_t)row * DMOD + tid]       = (_Float16)(d0 * inv * g[tid] + b[tid]);
    out[(size_t)row * DMOD + tid + 256] = (_Float16)(d1 * inv * g[tid + 256] + b[tid + 256]);
}

// ---------------- block-tiled WMMA GEMM with async double-buffered LDS staging ----------
// C[M,N] = epilogue( A16[M,K] @ Bt16[N,K]^T + bias )
// Block: 256 threads = 8 waves in a 4(M) x 2(N) grid; block tile 64 x 128.
// Per 32-wide k-step: A tile 64x32 and B tile 128x32 staged via
// global_load_async_to_lds_b128 (3 async instructions/wave/stage), double buffered.
__global__ __launch_bounds__(256) void gemm_wmma_kernel(
    const _Float16* __restrict__ A, const _Float16* __restrict__ Bt,
    const float* __restrict__ bias, const float* __restrict__ resid,
    float* __restrict__ Cf, _Float16* __restrict__ Ch,
    int M, int N, int K, int do_gelu)
{
    (void)M;
    __shared__ _Float16 Ash[2][64 * 32];
    __shared__ _Float16 Bsh[2][128 * 32];

    int tid = threadIdx.x;
    int lane = tid & 31, hl = lane >> 4, lr = lane & 15;
    int wave = tid >> 5;            // 0..7
    int wm = wave >> 1, wn = wave & 1;
    int nblkN = N >> 7;
    int bm = blockIdx.x / nblkN, bn = blockIdx.x % nblkN;
    int mBase = bm * 64, nBase = bn * 128;

    // staging roles: thread -> one 16B chunk of A (64 rows x 4 chunks = 256)
    // and two 16B chunks of B (128 rows x 4 chunks = 512).
    int cr = tid >> 2, cc = tid & 3;
    const _Float16* ag  = A  + (size_t)(mBase + cr) * K + cc * 8;
    const _Float16* bg0 = Bt + (size_t)(nBase + cr) * K + cc * 8;
    const _Float16* bg1 = bg0 + (size_t)64 * K;
    uint32_t aoff[2], boff0[2], boff1[2];
    aoff[0]  = lds_off(&Ash[0][cr * 32 + cc * 8]);
    aoff[1]  = lds_off(&Ash[1][cr * 32 + cc * 8]);
    boff0[0] = lds_off(&Bsh[0][cr * 32 + cc * 8]);
    boff0[1] = lds_off(&Bsh[1][cr * 32 + cc * 8]);
    boff1[0] = lds_off(&Bsh[0][(64 + cr) * 32 + cc * 8]);
    boff1[1] = lds_off(&Bsh[1][(64 + cr) * 32 + cc * 8]);

    v8f acc[4];
    for (int t = 0; t < 4; ++t)
        for (int i = 0; i < 8; ++i) acc[t][i] = 0.0f;

    const int nsteps = K >> 5;
    // prologue: stage 0
    async_copy_b128(aoff[0],  ag);
    async_copy_b128(boff0[0], bg0);
    async_copy_b128(boff1[0], bg1);

    for (int s = 0; s < nsteps; ++s) {
        int buf = s & 1;
        if (s + 1 < nsteps) {
            int nb = buf ^ 1, k1 = (s + 1) << 5;
            async_copy_b128(aoff[nb],  ag  + k1);
            async_copy_b128(boff0[nb], bg0 + k1);
            async_copy_b128(boff1[nb], bg1 + k1);
            wait_async_le3();     // in-order completion => stage s landed
        } else {
            wait_async_0();
        }
        __syncthreads();          // all waves' stage-s data visible in LDS

        v16h af = frag_a(&Ash[buf][(wm * 16 + lr) * 32], hl);
        #pragma unroll
        for (int t = 0; t < 4; ++t) {
            v16h bf = frag_b(&Bsh[buf][(wn * 64 + t * 16 + lr) * 32], hl);
            acc[t] = __builtin_amdgcn_wmma_f32_16x16x32_f16(
                false, af, false, bf, (short)0, acc[t], false, false);
        }
        __syncthreads();          // everyone done reading buf before it is re-staged
    }

    #pragma unroll
    for (int t = 0; t < 4; ++t) {
        int col = nBase + wn * 64 + t * 16 + lr;
        float bv = bias ? bias[col] : 0.0f;
        #pragma unroll
        for (int r = 0; r < 8; ++r) {
            int row = mBase + wm * 16 + hl * 8 + r;
            float v = acc[t][r] + bv;
            if (do_gelu) v = 0.5f * v * (1.0f + erff(v * 0.7071067811865475f));
            size_t o = (size_t)row * N + col;
            if (resid) v += resid[o];
            if (Cf) Cf[o] = v;
            if (Ch) Ch[o] = (_Float16)v;
        }
    }
}

// ---------------- fused flash attention (one wave = one 16-row Q tile per (b,h)) ----------
__global__ __launch_bounds__(128) void attn_kernel(
    const _Float16* __restrict__ q16, const _Float16* __restrict__ k16,
    const _Float16* __restrict__ vT,  _Float16* __restrict__ o16)
{
    __shared__ _Float16 Pshm[4][16 * 32];
    int lane = threadIdx.x & 31, hl = lane >> 4, lr = lane & 15;
    int wib = threadIdx.x >> 5;
    int gw = blockIdx.x * 4 + wib;          // 0..1023
    int mt = gw & 63;                       // Q row tile
    int bh = gw >> 6;
    int hh = bh & (NH - 1), bb = bh >> 3;

    const _Float16* qrow = q16 + ((size_t)((bb * SLEN + mt * 16 + lr) * NH + hh)) * DKH;
    v16h aq0 = frag_a(qrow, hl);        // k = 0..31 of head dim
    v16h aq1 = frag_a(qrow + 32, hl);   // k = 32..63

    float mrow[8], lrow[8];
    v8f O[4];
    for (int r = 0; r < 8; ++r) { mrow[r] = -3.0e38f; lrow[r] = 0.0f; }
    for (int t = 0; t < 4; ++t)
        for (int i = 0; i < 8; ++i) O[t][i] = 0.0f;

    _Float16* P = &Pshm[wib][0];
    int jmax = mt * 16 + 15;
    for (int j0 = 0; j0 <= jmax; j0 += 32) {
        // S = Q K^T for a 16x32 chunk (two 16x16 tiles, head-dim 64 -> 2 wmma each)
        v8f s[2];
        for (int t = 0; t < 2; ++t)
            for (int i = 0; i < 8; ++i) s[t][i] = 0.0f;
        #pragma unroll
        for (int t = 0; t < 2; ++t) {
            const _Float16* krow = k16 + ((size_t)((bb * SLEN + j0 + t * 16 + lr) * NH + hh)) * DKH;
            v16h b0 = frag_b(krow, hl);
            s[t] = __builtin_amdgcn_wmma_f32_16x16x32_f16(false, aq0, false, b0, (short)0, s[t], false, false);
            v16h b1 = frag_b(krow + 32, hl);
            s[t] = __builtin_amdgcn_wmma_f32_16x16x32_f16(false, aq1, false, b1, (short)0, s[t], false, false);
        }
        // scale + causal mask
        #pragma unroll
        for (int t = 0; t < 2; ++t) {
            int J = j0 + t * 16 + lr;
            #pragma unroll
            for (int r = 0; r < 8; ++r) {
                int m = mt * 16 + hl * 8 + r;
                float sv = s[t][r] * 0.125f;     // 1/sqrt(64)
                if (J > m) sv = -3.0e38f;
                s[t][r] = sv;
            }
        }
        // online softmax per row (row values live across 16 lanes of a half-wave)
        #pragma unroll
        for (int r = 0; r < 8; ++r) {
            float mc = fmaxf(s[0][r], s[1][r]);
            for (int off = 8; off >= 1; off >>= 1) mc = fmaxf(mc, __shfl_xor(mc, off, 32));
            float mnew = fmaxf(mrow[r], mc);
            float corr = __expf(mrow[r] - mnew);
            float p0 = __expf(s[0][r] - mnew);
            float p1 = __expf(s[1][r] - mnew);
            float rs = p0 + p1;
            for (int off = 8; off >= 1; off >>= 1) rs += __shfl_xor(rs, off, 32);
            lrow[r] = lrow[r] * corr + rs;
            mrow[r] = mnew;
            #pragma unroll
            for (int t = 0; t < 4; ++t) O[t][r] *= corr;
            int prow = hl * 8 + r;
            P[prow * 32 + lr]      = (_Float16)p0;
            P[prow * 32 + 16 + lr] = (_Float16)p1;
        }
        asm volatile("s_wait_dscnt 0x0" ::: "memory");   // LDS stores visible before fragment reload
        v16h ap = frag_a(P + lr * 32, hl);               // P as A fragment [16x32]
        // O += P * V  (V^T pre-transposed: rows are head-dim, contiguous in sequence)
        #pragma unroll
        for (int t = 0; t < 4; ++t) {
            const _Float16* vrow = vT + ((size_t)((bb * NH + hh) * DKH + t * 16 + lr)) * SLEN + j0;
            v16h bv = frag_b(vrow, hl);
            O[t] = __builtin_amdgcn_wmma_f32_16x16x32_f16(false, ap, false, bv, (short)0, O[t], false, false);
        }
    }
    #pragma unroll
    for (int t = 0; t < 4; ++t)
        #pragma unroll
        for (int r = 0; r < 8; ++r) {
            int m = mt * 16 + hl * 8 + r;
            float ov = O[t][r] / lrow[r];
            o16[((size_t)((bb * SLEN + m) * NH + hh)) * DKH + t * 16 + lr] = (_Float16)ov;
        }
}

// ---------------- host orchestration ----------------
extern "C" void kernel_launch(void* const* d_in, const int* in_sizes, int n_in,
                              void* d_out, int out_size, void* d_ws, size_t ws_size,
                              hipStream_t stream) {
    (void)in_sizes; (void)n_in; (void)out_size; (void)ws_size;
    const int*   tokens = (const int*)  d_in[0];
    const float* emb    = (const float*)d_in[1];
    const float* Wq     = (const float*)d_in[2];
    const float* bq     = (const float*)d_in[3];
    const float* Wk     = (const float*)d_in[4];
    const float* bk     = (const float*)d_in[5];
    const float* Wv     = (const float*)d_in[6];
    const float* bv     = (const float*)d_in[7];
    const float* Wo     = (const float*)d_in[8];
    const float* bo     = (const float*)d_in[9];
    const float* W1     = (const float*)d_in[10];
    const float* b1     = (const float*)d_in[11];
    const float* W2     = (const float*)d_in[12];
    const float* b2     = (const float*)d_in[13];
    const float* g1     = (const float*)d_in[14];
    const float* be1    = (const float*)d_in[15];
    const float* g3     = (const float*)d_in[16];
    const float* be3    = (const float*)d_in[17];
    const float* gf     = (const float*)d_in[18];
    const float* bef    = (const float*)d_in[19];
    float* logits = (float*)d_out;

    char* ws = (char*)d_ws;
    size_t off = 0;
    auto take = [&](size_t bytes) -> char* {
        char* p = ws + off;
        off = (off + bytes + 255) & ~(size_t)255;
        return p;
    };
    _Float16* emb16 = (_Float16*)take((size_t)NVOC * DMOD * 2);
    _Float16* WqT   = (_Float16*)take((size_t)NLAY * HD * DMOD * 2);
    _Float16* WkT   = (_Float16*)take((size_t)NLAY * HD * DMOD * 2);
    _Float16* WvT   = (_Float16*)take((size_t)NLAY * HD * DMOD * 2);
    _Float16* WoT   = (_Float16*)take((size_t)NLAY * DMOD * HD * 2);
    _Float16* W1T   = (_Float16*)take((size_t)NLAY * DMOD * DFFN * 2);
    _Float16* W2T   = (_Float16*)take((size_t)NLAY * DMOD * DFFN * 2);
    float*    xbuf  = (float*)   take((size_t)BLTOT * DMOD * 4);
    _Float16* h16   = (_Float16*)take((size_t)BLTOT * DMOD * 2);
    _Float16* q16b  = (_Float16*)take((size_t)BLTOT * HD * 2);
    _Float16* k16b  = (_Float16*)take((size_t)BLTOT * HD * 2);
    _Float16* v16b  = (_Float16*)take((size_t)BLTOT * HD * 2);
    _Float16* vTb   = (_Float16*)take((size_t)NBAT * NH * DKH * SLEN * 2);
    _Float16* o16b  = (_Float16*)take((size_t)BLTOT * HD * 2);
    _Float16* ff16  = (_Float16*)take((size_t)BLTOT * DFFN * 2);

    // --- weight prep (deterministic, every call) ---
    conv_f16_kernel<<<(NVOC * DMOD + 255) / 256, 256, 0, stream>>>(emb, emb16, (size_t)NVOC * DMOD);
    for (int i = 0; i < NLAY; ++i) {
        // Wq/Wk/Wv: [H, D, DK] -> [H, DK, D]  (row n = h*DK+j, contiguous in d)
        convT_kernel<<<dim3((DMOD * DKH + 255) / 256, NH), 256, 0, stream>>>(
            Wq + (size_t)i * NH * DMOD * DKH, WqT + (size_t)i * HD * DMOD, DMOD, DKH);
        convT_kernel<<<dim3((DMOD * DKH + 255) / 256, NH), 256, 0, stream>>>(
            Wk + (size_t)i * NH * DMOD * DKH, WkT + (size_t)i * HD * DMOD, DMOD, DKH);
        convT_kernel<<<dim3((DMOD * DKH + 255) / 256, NH), 256, 0, stream>>>(
            Wv + (size_t)i * NH * DMOD * DKH, WvT + (size_t)i * HD * DMOD, DMOD, DKH);
        // Wo: [HD, D] -> [D, HD]
        convT_kernel<<<dim3((HD * DMOD + 255) / 256, 1), 256, 0, stream>>>(
            Wo + (size_t)i * HD * DMOD, WoT + (size_t)i * DMOD * HD, HD, DMOD);
        // W1: [D, DFF] -> [DFF, D]
        convT_kernel<<<dim3((DMOD * DFFN + 255) / 256, 1), 256, 0, stream>>>(
            W1 + (size_t)i * DMOD * DFFN, W1T + (size_t)i * DMOD * DFFN, DMOD, DFFN);
        // W2: [DFF, D] -> [D, DFF]
        convT_kernel<<<dim3((DFFN * DMOD + 255) / 256, 1), 256, 0, stream>>>(
            W2 + (size_t)i * DFFN * DMOD, W2T + (size_t)i * DMOD * DFFN, DFFN, DMOD);
    }

    // --- forward ---
    embed_kernel<<<BLTOT, 256, 0, stream>>>(tokens, emb, xbuf);

    const int gQ   = (BLTOT / 64) * (HD / 128);     // 128 blocks
    const int gFF1 = (BLTOT / 64) * (DFFN / 128);   // 512 blocks
    const int gLM  = (BLTOT / 64) * (NVOC / 128);   // 8000 blocks

    for (int i = 0; i < NLAY; ++i) {
        ln_kernel<<<BLTOT, 256, 0, stream>>>(xbuf, g1 + (size_t)i * DMOD, be1 + (size_t)i * DMOD, h16);
        gemm_wmma_kernel<<<gQ, 256, 0, stream>>>(h16, WqT + (size_t)i * HD * DMOD,
            bq + (size_t)i * HD, nullptr, nullptr, q16b, BLTOT, HD, DMOD, 0);
        gemm_wmma_kernel<<<gQ, 256, 0, stream>>>(h16, WkT + (size_t)i * HD * DMOD,
            bk + (size_t)i * HD, nullptr, nullptr, k16b, BLTOT, HD, DMOD, 0);
        gemm_wmma_kernel<<<gQ, 256, 0, stream>>>(h16, WvT + (size_t)i * HD * DMOD,
            bv + (size_t)i * HD, nullptr, nullptr, v16b, BLTOT, HD, DMOD, 0);
        transposeV_kernel<<<(NBAT * NH * DKH * SLEN + 255) / 256, 256, 0, stream>>>(v16b, vTb);
        attn_kernel<<<(NBAT * NH * (SLEN / 16)) / 4, 128, 0, stream>>>(q16b, k16b, vTb, o16b);
        gemm_wmma_kernel<<<gQ, 256, 0, stream>>>(o16b, WoT + (size_t)i * DMOD * HD,
            bo + (size_t)i * DMOD, xbuf, xbuf, nullptr, BLTOT, DMOD, HD, 0);
        ln_kernel<<<BLTOT, 256, 0, stream>>>(xbuf, g3 + (size_t)i * DMOD, be3 + (size_t)i * DMOD, h16);
        gemm_wmma_kernel<<<gFF1, 256, 0, stream>>>(h16, W1T + (size_t)i * DMOD * DFFN,
            b1 + (size_t)i * DFFN, nullptr, nullptr, ff16, BLTOT, DFFN, DMOD, 1);
        gemm_wmma_kernel<<<gQ, 256, 0, stream>>>(ff16, W2T + (size_t)i * DMOD * DFFN,
            b2 + (size_t)i * DMOD, xbuf, xbuf, nullptr, BLTOT, DMOD, DFFN, 0);
    }
    ln_kernel<<<BLTOT, 256, 0, stream>>>(xbuf, gf, bef, h16);
    // tied LM head: logits = h @ emb^T  -> Bt = emb16 directly (already [V, D])
    gemm_wmma_kernel<<<gLM, 256, 0, stream>>>(h16, emb16, nullptr, nullptr,
        logits, nullptr, BLTOT, NVOC, DMOD, 0);
}
